// Multi_Attention_Layer_71992241815773
// MI455X (gfx1250) — compile-verified, matrix operand order checked by reading, not run on hardware
//
#include <hip/hip_runtime.h>
#include <math.h>

// ---------------------------------------------------------------------------
// MHA for MI455X (gfx1250, wave32, WMMA). All GEMMs via v_wmma_f32_16x16x32_f16.
// K/V tiles staged into LDS by the Tensor Data Mover when available.
// ---------------------------------------------------------------------------

typedef __attribute__((ext_vector_type(16))) _Float16 v16h;
typedef __attribute__((ext_vector_type(8)))  float    v8f;
typedef _Float16 half_t;

typedef __attribute__((ext_vector_type(4))) unsigned int u32x4;
typedef __attribute__((ext_vector_type(4))) int          i32x4;
typedef __attribute__((ext_vector_type(8))) int          i32x8;

#define EDIM   512
#define NHEADS 8
#define DH     64
#define LQ     256
#define SK     256
#define BATCH  128

#if __has_builtin(__builtin_amdgcn_tensor_load_to_lds)
#define HAVE_TDM 1
#if __has_include(<hip/amd_detail/amd_gfx1250_TDM.h>)
#define TDM_6ARG 1   // therock headers => 6-arg builtin
#endif
#endif

static __device__ __forceinline__ v8f zero8() {
  v8f z = {0.f, 0.f, 0.f, 0.f, 0.f, 0.f, 0.f, 0.f};
  return z;
}

static __device__ __forceinline__ v8f wmma16(v16h a, v16h b, v8f c) {
  // (neg_a, A, neg_b, B, c_mod, C, reuse_a, reuse_b)
  return __builtin_amdgcn_wmma_f32_16x16x32_f16(false, a, false, b, (short)0, c,
                                                false, false);
}

// 16-bit A/B fragment: lane&15 -> row (A) / col (B);
// element i -> K = (i<8?0:16) + 8*(lane>=16) + (i&7)  => two 8-half chunks.
static __device__ __forceinline__ v16h load_frag_h(const half_t* __restrict__ base,
                                                   int ld, int row, int k0, int lane) {
  const int lh8 = ((lane >> 4) & 1) * 8;
  const half_t* p = base + (size_t)row * ld + k0 + lh8;
  v16h f;
  ((uint4*)&f)[0] = *(const uint4*)(p);
  ((uint4*)&f)[1] = *(const uint4*)(p + 16);
  return f;
}

static __device__ __forceinline__ v16h load_frag_f32(const float* __restrict__ base,
                                                     int ld, int row, int k0, int lane) {
  const int lh8 = ((lane >> 4) & 1) * 8;
  const float* p = base + (size_t)row * ld + k0 + lh8;
  v16h f;
#pragma unroll
  for (int j = 0; j < 8; ++j) {
    f[j]     = (half_t)p[j];
    f[j + 8] = (half_t)p[16 + j];
  }
  return f;
}

// B fragment of V (K-dim = s). Vs is [s][d] with row stride 72 halves.
static __device__ __forceinline__ v16h load_vfrag(const half_t* __restrict__ Vs,
                                                  int s0, int dbase, int lane) {
  const int l15 = lane & 15;
  const int lh8 = ((lane >> 4) & 1) * 8;
  v16h f;
#pragma unroll
  for (int j = 0; j < 8; ++j) {
    f[j]     = Vs[(size_t)(s0 + lh8 + j) * 72 + dbase + l15];
    f[j + 8] = Vs[(size_t)(s0 + 16 + lh8 + j) * 72 + dbase + l15];
  }
  return f;
}

#if defined(HAVE_TDM)
// TDM: load a 256x64-half (S x Dh) tile, contiguous in global memory, into LDS
// at byte offset `lds_off`, inserting 4 DWORDs (16B) of padding after every
// 32 DWORDs (128B row) => LDS row stride 72 halves, matching the readers.
static __device__ __forceinline__ void tdm_load_kv_tile(unsigned long long gaddr,
                                                        unsigned lds_off) {
  u32x4 g0;
  g0[0] = 1u;                                            // count=1 (valid D#)
  g0[1] = lds_off;                                       // lds_addr
  g0[2] = (unsigned)(gaddr & 0xFFFFFFFFull);             // global_addr[31:0]
  g0[3] = (unsigned)((gaddr >> 32) & 0x1FFFFFFull)       // global_addr[56:32]
        | (2u << 30);                                    // type=2 ("image")
  i32x8 g1;
  g1[0] = (1 << 16)            // data_size = 1 -> 2-byte elements
        | (1 << 20)            // pad_enable
        | (4 << 22)            // pad_interval: code 4 -> every 32 DWORDs
        | (3 << 25);           // pad_amount:  code 3 -> 4 DWORDs (16B)
  g1[1] = (DH & 0xFFFF) << 16;                  // tensor_dim0 = 64 (low16)
  g1[2] = (SK & 0xFFFF) << 16;                  // dim0 hi=0 | tensor_dim1 = 256
  g1[3] = (DH & 0xFFFF) << 16;                  // dim1 hi=0 | tile_dim0 = 64
  g1[4] = SK;                                   // tile_dim1 = 256 | tile_dim2 = 0
  g1[5] = DH;                                   // tensor_dim0_stride = 64
  g1[6] = (int)(((DH * SK) & 0xFFFF) << 16);    // stride0 hi=0 | stride1 lo
  g1[7] = (DH * SK) >> 16;                      // stride1 hi
  i32x4 gz = {0, 0, 0, 0};
#if defined(TDM_6ARG)
  i32x8 gz8 = {0, 0, 0, 0, 0, 0, 0, 0};
  __builtin_amdgcn_tensor_load_to_lds(g0, g1, gz, gz, gz8, 0);
#else
  __builtin_amdgcn_tensor_load_to_lds(g0, g1, gz, gz, 0);
#endif
}

static __device__ __forceinline__ void tdm_wait0() {
#if __has_builtin(__builtin_amdgcn_s_wait_tensorcnt)
  __builtin_amdgcn_s_wait_tensorcnt(0);
#else
  asm volatile("s_wait_tensorcnt 0x0" ::: "memory");
#endif
}
#endif  // HAVE_TDM

// ---------------------------------------------------------------------------
// Weight conversion fp32 -> f16 (weights then live in L2, re-read cheaply)
// ---------------------------------------------------------------------------
__global__ __launch_bounds__(256) void convert_weights(
    const float* __restrict__ win, const float* __restrict__ wout,
    half_t* __restrict__ W16in, half_t* __restrict__ W16out) {
  const int idx = blockIdx.x * 256 + threadIdx.x;
  if (idx < 3 * EDIM * EDIM) W16in[idx] = (half_t)win[idx];
  if (idx < EDIM * EDIM)     W16out[idx] = (half_t)wout[idx];
}

// ---------------------------------------------------------------------------
// key_padding_mask[b][s] = all(neighbors[s,b,:] == 0) -> -inf else 0
// ---------------------------------------------------------------------------
__global__ __launch_bounds__(256) void build_mask(const float* __restrict__ nb,
                                                  float* __restrict__ maskf) {
  const int idx = blockIdx.x * 256 + threadIdx.x;  // idx = s*BATCH + b
  const float* p = nb + (size_t)idx * EDIM;
  int nz = 0;
#pragma unroll 4
  for (int e = 0; e < EDIM; e += 4) {
    float4 v = *(const float4*)(p + e);
    nz |= (v.x != 0.f) | (v.y != 0.f) | (v.z != 0.f) | (v.w != 0.f);
  }
  const int s = idx >> 7;          // /BATCH
  const int b = idx & (BATCH - 1);
  maskf[b * SK + s] = nz ? 0.0f : -INFINITY;
}

// ---------------------------------------------------------------------------
// Generic projection GEMM:  dst[m,n] = sum_k src[m,k] * W[n,k] + bias[n]
// Block tile 128(M) x 64(N), 8 waves each owning 16x64, K-step 32.
// DST_F32==0: store f16 into [B][H][pos][Dh];  DST_F32==1: fp32 row-major.
// ---------------------------------------------------------------------------
template <int DST_F32, int SRC_HALF>
__global__ __launch_bounds__(256) void gemm_proj(
    const void* __restrict__ srcv, const half_t* __restrict__ W16,
    const float* __restrict__ bias, void* __restrict__ dstv) {
  __shared__ half_t Bs[64 * 40];  // W tile [n][k], stride 40 halves (16B aligned)

  const int tid  = threadIdx.x;
  const int lane = tid & 31, wave = tid >> 5;
  const int l15  = lane & 15, lh = (lane >> 4) & 1;
  const int m0 = blockIdx.x * 128, n0 = blockIdx.y * 64;
  const int mrow = m0 + wave * 16 + l15;

  v8f acc[4];
#pragma unroll
  for (int t = 0; t < 4; ++t) acc[t] = zero8();

  const int bn = tid >> 2;        // 0..63  (tile col n)
  const int bk = (tid & 3) * 8;   // 0,8,16,24

  for (int k0 = 0; k0 < EDIM; k0 += 32) {
    // stage W tile (f16) into LDS
    *(uint4*)(Bs + bn * 40 + bk) =
        *(const uint4*)(W16 + (size_t)(n0 + bn) * EDIM + k0 + bk);
    if (k0 + 32 < EDIM)  // -> global_prefetch_b8 (probe-verified lowering)
      __builtin_prefetch(W16 + (size_t)(n0 + bn) * EDIM + k0 + 32 + bk, 0, 1);
    __syncthreads();

    v16h a = SRC_HALF ? load_frag_h((const half_t*)srcv, EDIM, mrow, k0, lane)
                      : load_frag_f32((const float*)srcv, EDIM, mrow, k0, lane);
#pragma unroll
    for (int t = 0; t < 4; ++t) {
      v16h bfrag = load_frag_h(Bs, 40, t * 16 + l15, 0, lane);
      acc[t] = wmma16(a, bfrag, acc[t]);
    }
    __syncthreads();
  }

#pragma unroll
  for (int t = 0; t < 4; ++t) {
#pragma unroll
    for (int r = 0; r < 8; ++r) {
      const int m = m0 + wave * 16 + r + lh * 8;  // C layout: VGPR r -> row r+8*lh
      const int n = n0 + t * 16 + l15;            // N = lane&15
      float v = acc[t][r] + bias[n];
      if (DST_F32) {
        ((float*)dstv)[(size_t)m * EDIM + n] = v;
      } else {
        const int bb  = m & (BATCH - 1);  // m = pos*BATCH + b
        const int pos = m >> 7;
        const int hh  = n >> 6;           // n = h*64 + d
        const int d   = n & (DH - 1);
        ((half_t*)dstv)[((size_t)(bb * NHEADS + hh) * SK + pos) * DH + d] = (half_t)v;
      }
    }
  }
}

// ---------------------------------------------------------------------------
// Flash-style attention, one (b,h) + 128 L-rows per block; 16 rows per wave.
// K/V (256x64 f16 each) staged to LDS by the TDM (padded to 72-half stride,
// 16B-aligned rows, conflict-free fragment gathers). Online softmax; P is
// re-fed through a per-wave LDS scratch to become a WMMA A-fragment.
// ---------------------------------------------------------------------------
__global__ __launch_bounds__(256) void attn_kernel(
    const half_t* __restrict__ Q, const half_t* __restrict__ K,
    const half_t* __restrict__ V, const float* __restrict__ maskf,
    half_t* __restrict__ ctx) {
  extern __shared__ char smem_raw[];
  half_t* Ks = (half_t*)smem_raw;   // [256][72]
  half_t* Vs = Ks + SK * 72;        // [256][72]
  half_t* Ps = Vs + SK * 72;        // [8 waves][16][40]

  const int tid  = threadIdx.x;
  const int lane = tid & 31, wave = tid >> 5;
  const int l15  = lane & 15, lh = (lane >> 4) & 1;
  const int bh = blockIdx.x;        // b*NHEADS + h
  const int b  = bh >> 3, h = bh & 7;
  const int l0 = blockIdx.y * 128 + wave * 16;

  const half_t* Kbh = K + (size_t)bh * SK * DH;
  const half_t* Vbh = V + (size_t)bh * SK * DH;
  const half_t* Qbh = Q + (size_t)bh * LQ * DH;

#if defined(HAVE_TDM)
  if (wave == 0) {  // one wave drives the Tensor Data Mover for both tiles
    tdm_load_kv_tile((unsigned long long)(size_t)Kbh, 0u);
    tdm_load_kv_tile((unsigned long long)(size_t)Vbh, (unsigned)(SK * 72 * 2));
    tdm_wait0();    // TENSORcnt -> 0 before publishing via the barrier
  }
  __syncthreads();
#else
  {  // fallback: cooperative K/V stage, one row of 64 halves per thread
    const uint4* gk = (const uint4*)(Kbh + (size_t)tid * DH);
    const uint4* gv = (const uint4*)(Vbh + (size_t)tid * DH);
    uint4* lk = (uint4*)(Ks + (size_t)tid * 72);
    uint4* lv = (uint4*)(Vs + (size_t)tid * 72);
#pragma unroll
    for (int j = 0; j < 8; ++j) { lk[j] = gk[j]; lv[j] = gv[j]; }
  }
  __syncthreads();
#endif

  // Q fragment (16 rows x Dh=64) held in registers for whole S loop
  const v16h qf0 = load_frag_h(Qbh, DH, l0 + l15, 0, lane);
  const v16h qf1 = load_frag_h(Qbh, DH, l0 + l15, 32, lane);

  v8f cacc[4];
#pragma unroll
  for (int t = 0; t < 4; ++t) cacc[t] = zero8();
  float rm[8], rl[8];
#pragma unroll
  for (int r = 0; r < 8; ++r) { rm[r] = -1e30f; rl[r] = 0.f; }

  half_t* Pw = Ps + wave * (16 * 40);
  const float scale = 0.125f;  // 1/sqrt(Dh)

  for (int s0 = 0; s0 < SK; s0 += 32) {
    // scores: Q(16x64) @ K^T(64x32) -> two 16x16 tiles
    v8f sc0 = zero8(), sc1 = zero8();
    {
      v16h kb0 = load_frag_h(Ks, 72, s0 + l15, 0, lane);
      v16h kb1 = load_frag_h(Ks, 72, s0 + l15, 32, lane);
      sc0 = wmma16(qf0, kb0, sc0);
      sc0 = wmma16(qf1, kb1, sc0);
      v16h kb2 = load_frag_h(Ks, 72, s0 + 16 + l15, 0, lane);
      v16h kb3 = load_frag_h(Ks, 72, s0 + 16 + l15, 32, lane);
      sc1 = wmma16(qf0, kb2, sc1);
      sc1 = wmma16(qf1, kb3, sc1);
    }
    const float mk0 = maskf[b * SK + s0 + l15];
    const float mk1 = maskf[b * SK + s0 + 16 + l15];

    float mnew[8], fac[8];
#pragma unroll
    for (int r = 0; r < 8; ++r) {
      float a0 = sc0[r] * scale + mk0;
      float a1 = sc1[r] * scale + mk1;
      sc0[r] = a0; sc1[r] = a1;
      float mx = fmaxf(a0, a1);
      mx = fmaxf(mx, __shfl_xor(mx, 1, 16));
      mx = fmaxf(mx, __shfl_xor(mx, 2, 16));
      mx = fmaxf(mx, __shfl_xor(mx, 4, 16));
      mx = fmaxf(mx, __shfl_xor(mx, 8, 16));
      mnew[r] = fmaxf(rm[r], mx);
      fac[r]  = __expf(rm[r] - mnew[r]);
      rm[r]   = mnew[r];
    }
#pragma unroll
    for (int r = 0; r < 8; ++r) {
      float p0 = __expf(sc0[r] - mnew[r]);
      float p1 = __expf(sc1[r] - mnew[r]);
      float ps = p0 + p1;
      ps += __shfl_xor(ps, 1, 16);
      ps += __shfl_xor(ps, 2, 16);
      ps += __shfl_xor(ps, 4, 16);
      ps += __shfl_xor(ps, 8, 16);
      rl[r] = rl[r] * fac[r] + ps;
      const int prow = r + lh * 8;  // C layout -> P row index
      Pw[prow * 40 + l15]      = (half_t)p0;
      Pw[prow * 40 + 16 + l15] = (half_t)p1;
#pragma unroll
      for (int t = 0; t < 4; ++t) cacc[t][r] *= fac[r];  // rescale ctx
    }
    // cross-lane LDS RAW within the wave: wait for DS writes to land
    asm volatile("s_wait_dscnt 0x0" ::: "memory");

    const v16h pf = load_frag_h(Pw, 40, l15, 0, lane);  // P as A-fragment
#pragma unroll
    for (int t = 0; t < 4; ++t) {
      v16h vb = load_vfrag(Vs, s0, t * 16, lane);
      cacc[t] = wmma16(pf, vb, cacc[t]);
    }
  }

  // ctx[l][b][h*64+d] = acc / rowsum
#pragma unroll
  for (int t = 0; t < 4; ++t) {
#pragma unroll
    for (int r = 0; r < 8; ++r) {
      const int l = l0 + r + lh * 8;
      const int e = h * DH + t * 16 + l15;
      ctx[((size_t)l * BATCH + b) * EDIM + e] = (half_t)(cacc[t][r] / rl[r]);
    }
  }
}

// ---------------------------------------------------------------------------
extern "C" void kernel_launch(void* const* d_in, const int* in_sizes, int n_in,
                              void* d_out, int out_size, void* d_ws, size_t ws_size,
                              hipStream_t stream) {
  (void)in_sizes; (void)n_in; (void)out_size; (void)ws_size;
  const float* x    = (const float*)d_in[0];
  const float* nb   = (const float*)d_in[1];
  const float* win  = (const float*)d_in[2];
  const float* bin  = (const float*)d_in[3];
  const float* wout = (const float*)d_in[4];
  const float* bout = (const float*)d_in[5];

  // workspace layout (f16 unless noted); ~136 MB total
  half_t* W16in  = (half_t*)d_ws;                                   // 1536*512
  half_t* W16out = W16in + (size_t)3 * EDIM * EDIM;                 // 512*512
  half_t* Q16    = W16out + (size_t)EDIM * EDIM;                    // [B][H][L][Dh]
  half_t* K16    = Q16 + (size_t)BATCH * NHEADS * LQ * DH;          // [B][H][S][Dh]
  half_t* V16    = K16 + (size_t)BATCH * NHEADS * SK * DH;          // [B][H][S][Dh]
  half_t* CTX    = V16 + (size_t)BATCH * NHEADS * SK * DH;          // [L][B][E]
  float*  maskf  = (float*)(CTX + (size_t)LQ * BATCH * EDIM);       // [B][S] fp32

  convert_weights<<<3 * EDIM * EDIM / 256, 256, 0, stream>>>(win, wout, W16in, W16out);
  build_mask<<<SK * BATCH / 256, 256, 0, stream>>>(nb, maskf);

  // Q = x@Wq^T, K = nb@Wk^T, V = nb@Wv^T  (M=32768, N=K=512)
  gemm_proj<0, 0><<<dim3(256, 8), 256, 0, stream>>>(x,  W16in,                   bin,            Q16);
  gemm_proj<0, 0><<<dim3(256, 8), 256, 0, stream>>>(nb, W16in + EDIM * EDIM,     bin + EDIM,     K16);
  gemm_proj<0, 0><<<dim3(256, 8), 256, 0, stream>>>(nb, W16in + 2 * EDIM * EDIM, bin + 2 * EDIM, V16);

  const size_t attn_lds = (size_t)(2 * SK * 72 + 8 * 16 * 40) * sizeof(half_t);  // 83968 B
  attn_kernel<<<dim3(BATCH * NHEADS, LQ / 128), 256, attn_lds, stream>>>(
      Q16, K16, V16, maskf, CTX);

  // out = ctx@Wout^T + bout  (fp32 output)
  gemm_proj<1, 1><<<dim3(256, 8), 256, 0, stream>>>(CTX, W16out, bout, d_out);
}
